// ValueCompressor_33071248179490
// MI455X (gfx1250) — matrix operand order repaired and checked by reading, not compiled, not run on hardware
//
#include <hip/hip_runtime.h>
#include <hip/hip_bf16.h>
#include <math.h>

#define FEAT   128
#define NE     128
#define INTER  512
#define TOKB   64
#define EPSV   1e-6f
#define LNEPS  1e-5f

#define XN_STR 136   // bf16 elems per row (128 + pad) -> 4-bank rotation
#define H_STR  520   // bf16 elems per row (512 + pad)
#define CD_STR 272   // bf16 elems per row (256 + pad)
#define RC_STR 132   // f32 elems per row

#define XN_BYTES (TOKB * XN_STR * 2)          // 17408
#define H_BYTES  (TOKB * H_STR * 2)           // 66560
#define U_BYTES  (XN_BYTES + H_BYTES)         // 83968 (also covers code 34816 / rec 33792)

typedef __attribute__((ext_vector_type(16))) __bf16 v16bf;
typedef __attribute__((ext_vector_type(8)))  float  v8f;

struct __align__(16) U4 { unsigned int x, y, z, w; };
union Frag { struct { U4 lo, hi; } u; v16bf v; };

__device__ __forceinline__ unsigned short f2bfbits(float f) {
  unsigned u = __builtin_bit_cast(unsigned, f);
  unsigned r = u + 0x7FFFu + ((u >> 16) & 1u);   // round-to-nearest-even
  return (unsigned short)(r >> 16);
}

// A fragment (16x32 bf16): lane holds row (lane%16); K chunks [off,off+8) and [off+16,off+24),
// off = 0 (lanes 0-15) or 8 (lanes 16-31).
__device__ __forceinline__ v16bf loadA_lds(const unsigned short* buf, int row, int stride,
                                           int kbase, int lhalf) {
  const unsigned short* p = buf + row * stride + kbase + (lhalf ? 8 : 0);
  Frag f;
  f.u.lo = *(const U4*)p;
  f.u.hi = *(const U4*)(p + 16);
  return f.v;
}

// B fragment (32x16 bf16) for D = A * W^T: column n of B = row n of W; lane (n0+lane%16)
// reads K chunks [off,off+8) and [off+16,off+24), off = kbase + (lane<16 ? 0 : 8).
__device__ __forceinline__ v16bf loadB_row(const unsigned short* w, int row, int kdim,
                                           int kbase, int lhalf) {
  const unsigned short* p = w + (size_t)row * kdim + kbase + (lhalf ? 8 : 0);
  Frag f;
  f.u.lo = *(const U4*)p;
  f.u.hi = *(const U4*)(p + 16);
  return f.v;
}

__device__ __forceinline__ v8f wmma_bf16(v16bf a, v16bf b, v8f c) {
  return __builtin_amdgcn_wmma_f32_16x16x32_bf16(false, a, false, b, (short)0, c, false, false);
}

// ---------------- prep kernels: f32 weights -> bf16 in workspace ----------------
__global__ void conv_bf16(const float* __restrict__ src, unsigned short* __restrict__ dst, int n) {
  int i = blockIdx.x * blockDim.x + threadIdx.x;
  if (i < n) dst[i] = f2bfbits(src[i]);
}

// Et[f*256 + c] = embedding[c*128 + f]  (transposed so decode-B rows are contiguous)
__global__ void emb_transpose(const float* __restrict__ e, unsigned short* __restrict__ et) {
  int i = blockIdx.x * blockDim.x + threadIdx.x;
  if (i < 128 * 256) {
    int f = i >> 8, c = i & 255;
    et[i] = f2bfbits(e[c * 128 + f]);
  }
}

// ---------------- fused main kernel: 64 tokens per block, 8 waves ----------------
__global__ __launch_bounds__(256) void vc_main(
    const float* __restrict__ x, const float* __restrict__ gum,
    const float* __restrict__ lnw, const float* __restrict__ lnb,
    const float* __restrict__ b1, const float* __restrict__ b2,
    const float* __restrict__ scale,
    const unsigned short* __restrict__ w1b, const unsigned short* __restrict__ w2b,
    const unsigned short* __restrict__ Et, float* __restrict__ out) {

  // Overlaid LDS: [xn | h] -> code -> rec
  __shared__ __align__(16) unsigned char sU[U_BYTES];
  __shared__ float sPre[TOKB];
  __shared__ float sFac[TOKB];
  unsigned short* sXn   = (unsigned short*)sU;               // 64 x 136 bf16
  unsigned short* sH    = (unsigned short*)(sU + XN_BYTES);  // 64 x 520 bf16
  unsigned short* sCode = (unsigned short*)sU;               // 64 x 272 bf16 (after h dead)
  float*          recf  = (float*)sU;                        // 64 x 132 f32  (after code dead)

  const int tid   = threadIdx.x;
  const int lane  = tid & 31;
  const int wv    = tid >> 5;          // 0..7
  const int lmod  = lane & 15;
  const int lhalf = lane >> 4;
  const long tok0 = (long)blockIdx.x * TOKB;

  // ---- phase 1: scale-div, prescale-normalize, LayerNorm -> xn bf16 ----
#pragma unroll
  for (int it = 0; it < 4; it++) {
    const int t  = it * 16 + (tid >> 4);   // token 0..63 (16 threads per token)
    const int f0 = (tid & 15) * 8;
    const float* px = x + (tok0 + t) * FEAT + f0;
    float v[8], sc[8];
#pragma unroll
    for (int j = 0; j < 8; j++) { sc[j] = scale[f0 + j]; v[j] = px[j] / (sc[j] + EPSV); }
    float s2 = 0.f;
#pragma unroll
    for (int j = 0; j < 8; j++) s2 += v[j] * v[j];
#pragma unroll
    for (int m = 8; m >= 1; m >>= 1) s2 += __shfl_xor(s2, m, 16);
    float pre = sqrtf(s2);
    if ((tid & 15) == 0) sPre[t] = pre;
    float inv = 1.f / (pre + EPSV);
#pragma unroll
    for (int j = 0; j < 8; j++) v[j] *= inv;
    float s = 0.f;
#pragma unroll
    for (int j = 0; j < 8; j++) s += v[j];
#pragma unroll
    for (int m = 8; m >= 1; m >>= 1) s += __shfl_xor(s, m, 16);
    float mu = s * (1.f / 128.f);
    float vv = 0.f;
#pragma unroll
    for (int j = 0; j < 8; j++) { float d = v[j] - mu; vv += d * d; }
#pragma unroll
    for (int m = 8; m >= 1; m >>= 1) vv += __shfl_xor(vv, m, 16);
    float rstd = rsqrtf(vv * (1.f / 128.f) + LNEPS);
#pragma unroll
    for (int j = 0; j < 8; j++) {
      float r = (v[j] - mu) * rstd * lnw[f0 + j] + lnb[f0 + j];
      sXn[t * XN_STR + f0 + j] = f2bfbits(r);
    }
  }
  __syncthreads();

  // ---- phase 2: GEMM1  h = gelu(xn @ w1^T + b1)  [64,128]x[128,512] ----
  {
    v8f acc[4][4] = {};
#pragma unroll
    for (int kk = 0; kk < 4; kk++) {
      v16bf a[4];
#pragma unroll
      for (int mt = 0; mt < 4; mt++)
        a[mt] = loadA_lds(sXn, mt * 16 + lmod, XN_STR, kk * 32, lhalf);
#pragma unroll
      for (int nt = 0; nt < 4; nt++) {
        v16bf b = loadB_row(w1b, wv * 64 + nt * 16 + lmod, FEAT, kk * 32, lhalf);
#pragma unroll
        for (int mt = 0; mt < 4; mt++)
          acc[mt][nt] = wmma_bf16(a[mt], b, acc[mt][nt]);
      }
    }
    // epilogue: bias + exact gelu -> sH (disjoint from sXn, no barrier needed)
#pragma unroll
    for (int nt = 0; nt < 4; nt++) {
      const int n = wv * 64 + nt * 16 + lmod;
      const float bias = b1[n];
#pragma unroll
      for (int mt = 0; mt < 4; mt++) {
#pragma unroll
        for (int r = 0; r < 8; r++) {
          float g = acc[mt][nt][r] + bias;
          g = 0.5f * g * (1.f + erff(g * 0.70710678118654752f));
          sH[(mt * 16 + r + lhalf * 8) * H_STR + n] = f2bfbits(g);
        }
      }
    }
  }
  __syncthreads();

  // ---- phase 3: GEMM2 + in-register gumbel hard argmax -> code bits ----
  {
    v8f acc[4][4] = {};
#pragma unroll 4
    for (int kk = 0; kk < 16; kk++) {
      v16bf a[4];
#pragma unroll
      for (int mt = 0; mt < 4; mt++)
        a[mt] = loadA_lds(sH, mt * 16 + lmod, H_STR, kk * 32, lhalf);
#pragma unroll
      for (int nt = 0; nt < 4; nt++) {
        v16bf b = loadB_row(w2b, wv * 64 + nt * 16 + lmod, INTER, kk * 32, lhalf);
#pragma unroll
        for (int mt = 0; mt < 4; mt++)
          acc[mt][nt] = wmma_bf16(a[mt], b, acc[mt][nt]);
      }
    }
    __syncthreads();   // all h reads done; code region may now overwrite xn/h

    // Epilogue: logits z = acc + b2 + gumbel; argmax over the 4-column group held by
    // 4 adjacent lanes (shfl_xor 1,2); first-max tie-break = lowest class index.
#pragma unroll
    for (int nt = 0; nt < 4; nt++) {
      const int n   = wv * 64 + nt * 16 + lmod;   // logits column (== flat gumbel idx)
      const int e   = n >> 2;                     // group
      const int cls = n & 3;                      // class within group
      const float bias = b2[n];
#pragma unroll
      for (int mt = 0; mt < 4; mt++) {
#pragma unroll
        for (int r = 0; r < 8; r++) {
          const int m = mt * 16 + r + lhalf * 8;  // token row
          const float u = gum[(tok0 + m) * 512 + n];
          float z = acc[mt][nt][r] + bias - logf(-logf(u));
          float zm = z; int cm = cls;
          {
            float oz = __shfl_xor(zm, 1); int oc = __shfl_xor(cm, 1);
            if (oz > zm || (oz == zm && oc < cm)) { zm = oz; cm = oc; }
          }
          {
            float oz = __shfl_xor(zm, 2); int oc = __shfl_xor(cm, 2);
            if (oz > zm || (oz == zm && oc < cm)) { zm = oz; cm = oc; }
          }
          // code column c = 2e + bit; lane cls==0 writes bit0, cls==1 writes bit1
          if (cls < 2) {
            unsigned short bit = ((cm == cls) || (cm == 2)) ? (unsigned short)0x3F80u
                                                            : (unsigned short)0;
            sCode[m * CD_STR + 2 * e + cls] = bit;
          }
        }
      }
    }
  }
  __syncthreads();

  // ---- phase 4: decode  rec = code @ embedding  [64,256]x[256,128] ----
  {
    v8f accd[4] = {};
#pragma unroll
    for (int kk = 0; kk < 8; kk++) {
      v16bf b = loadB_row(Et, wv * 16 + lmod, 256, kk * 32, lhalf);
#pragma unroll
      for (int mt = 0; mt < 4; mt++) {
        v16bf a = loadA_lds(sCode, mt * 16 + lmod, CD_STR, kk * 32, lhalf);
        accd[mt] = wmma_bf16(a, b, accd[mt]);
      }
    }
    __syncthreads();   // all code reads done before rec overwrites the region
#pragma unroll
    for (int mt = 0; mt < 4; mt++) {
#pragma unroll
      for (int r = 0; r < 8; r++)
        recf[(mt * 16 + r + lhalf * 8) * RC_STR + wv * 16 + lmod] = accd[mt][r];
    }
  }
  __syncthreads();

  // ---- phase 5: postscale + store ----
  if (tid < TOKB) {
    float s = 0.f;
    for (int f = 0; f < FEAT; f++) { float r = recf[tid * RC_STR + f]; s += r * r; }
    float post = sqrtf(s);
    sFac[tid] = (sPre[tid] + EPSV) / (post + EPSV);
  }
  __syncthreads();
#pragma unroll
  for (int it = 0; it < 4; it++) {
    const int t  = it * 16 + (tid >> 4);
    const int f0 = (tid & 15) * 8;
    float fac = sFac[t];
    float* po = out + (tok0 + t) * FEAT + f0;
#pragma unroll
    for (int j = 0; j < 8; j++)
      po[j] = recf[t * RC_STR + f0 + j] * fac * (scale[f0 + j] + EPSV);
  }
}

extern "C" void kernel_launch(void* const* d_in, const int* in_sizes, int n_in,
                              void* d_out, int out_size, void* d_ws, size_t ws_size,
                              hipStream_t stream) {
  const float* x     = (const float*)d_in[0];
  const float* gum   = (const float*)d_in[1];
  const float* lnw   = (const float*)d_in[2];
  const float* lnb   = (const float*)d_in[3];
  const float* w1    = (const float*)d_in[4];
  const float* b1    = (const float*)d_in[5];
  const float* w2    = (const float*)d_in[6];
  const float* b2    = (const float*)d_in[7];
  const float* emb   = (const float*)d_in[8];
  const float* scale = (const float*)d_in[9];
  float* out = (float*)d_out;

  unsigned short* w1b = (unsigned short*)d_ws;            // 512*128  = 65536 elems
  unsigned short* w2b = w1b + 512 * 128;                  // 512*512  = 262144 elems
  unsigned short* Et  = w2b + 512 * 512;                  // 128*256  = 32768 elems

  conv_bf16<<<(512 * 128 + 255) / 256, 256, 0, stream>>>(w1, w1b, 512 * 128);
  conv_bf16<<<(512 * 512 + 255) / 256, 256, 0, stream>>>(w2, w2b, 512 * 512);
  emb_transpose<<<(128 * 256 + 255) / 256, 256, 0, stream>>>(emb, Et);

  const int T = 8 * 8192;                                 // 65536 tokens
  vc_main<<<T / TOKB, 256, 0, stream>>>(x, gum, lnw, lnb, b1, b2, scale,
                                        w1b, w2b, Et, out);
}
